// SCD_cell_11879879544318
// MI455X (gfx1250) — matrix operand order repaired and checked by reading, not compiled, not run on hardware
//
#include <hip/hip_runtime.h>
#include <hip/hip_bf16.h>
#include <math.h>

typedef _Float16 f16;
typedef __attribute__((ext_vector_type(16))) _Float16 v16h;
typedef __attribute__((ext_vector_type(8)))  float    v8f;

#define HID   512
#define BATCH 8192
#define KTOT  1536          // 3 * 512
#define MROWS 2048          // 4 * H (row 2048 handled separately)
#define NKT   48            // KTOT / 32
#define NMT   128           // MROWS / 16

// ---------------------------------------------------------------------------
// Kernel 1: pack [U11 | U21 | W01] (f32, row-major 2049x512 each) into f16
// WMMA A-fragment order. Fragment (mt,kt) = 512 halfs; half index l*16+j maps
// to m = mt*16 + (l&15), k = kt*32 + j + (l>>4)*16
// ---------------------------------------------------------------------------
__global__ void pack_weights(const float* __restrict__ U11,
                             const float* __restrict__ U21,
                             const float* __restrict__ W01,
                             f16* __restrict__ Apk) {
    int idx  = blockIdx.x * 256 + threadIdx.x;   // 0 .. 3145727
    int j    = idx & 15;
    int l    = (idx >> 4) & 31;
    int frag = idx >> 9;                         // mt*NKT + kt
    int kt   = frag % NKT;
    int mt   = frag / NKT;
    int m    = mt * 16 + (l & 15);
    int k    = kt * 32 + j + ((l >> 4) << 4);
    float v;
    if (k < 512)       v = U11[m * 512 + k];
    else if (k < 1024) v = U21[m * 512 + (k - 512)];
    else               v = W01[m * 512 + (k - 1024)];
    Apk[idx] = (f16)v;
}

// ---------------------------------------------------------------------------
// Kernel 2: pre-scaled activation matrix Xs, stored TRANSPOSED (b-major,
// Xs[b*1536 + k], f16) so WMMA B-fragment loads are contiguous 32B per lane.
// ---------------------------------------------------------------------------
__global__ void prescale_transpose(const float* __restrict__ h,
                                   const float* __restrict__ ht,
                                   const float* __restrict__ hb,
                                   const float* __restrict__ z,
                                   const float* __restrict__ zb,
                                   f16* __restrict__ Xs) {
    __shared__ f16 tile[32][33];
    int tx = threadIdx.x & 31;
    int ty = threadIdx.x >> 5;
    int k0 = blockIdx.x * 32;
    int b0 = blockIdx.y * 32;
    int b  = b0 + tx;
    float zc  = z[b];
    float zbc = zb[b];
#pragma unroll
    for (int i = 0; i < 4; ++i) {
        int k = k0 + ty + i * 8;
        float v;
        if (k < 512)       v = (1.0f - zc) * h[(size_t)k * BATCH + b];
        else if (k < 1024) v = zc          * ht[(size_t)(k - 512) * BATCH + b];
        else               v = zbc         * hb[(size_t)(k - 1024) * BATCH + b];
        tile[ty + i * 8][tx] = (f16)v;
    }
    __syncthreads();
#pragma unroll
    for (int i = 0; i < 4; ++i) {
        int r  = ty + i * 8;
        int bo = b0 + r;
        Xs[(size_t)bo * KTOT + k0 + tx] = tile[tx][r];
    }
}

__device__ __forceinline__ float sigm(float x) {
    return 1.0f / (1.0f + __expf(-x));
}
__device__ __forceinline__ float tanh_fast(float x) {
    return 2.0f / (1.0f + __expf(-2.0f * x)) - 1.0f;
}

// CDNA5 async copy: global -> LDS, ASYNCcnt-tracked (GVS addressing mode).
__device__ __forceinline__ void async_copy_b128(uint32_t lds_addr,
                                                uint64_t saddr,
                                                uint32_t vofs) {
    asm volatile("global_load_async_to_lds_b128 %0, %1, %2"
                 :: "v"(lds_addr), "v"(vofs), "s"(saddr) : "memory");
}

// ---------------------------------------------------------------------------
// Kernel 3: fused GEMM + LSTM epilogue.
// grid.x = 64 column blocks (128 cols, wave w owns 16), grid.y = 8 t-groups
// (4 M-subtiles each). Per K-step: 16 shared A frags (4t x 4gates, 16 KB)
// streamed into LDS via global_load_async_to_lds_b128 (double buffered,
// s_wait_asynccnt + barrier). sched_group_barrier prescribes a 3-fragment
// lookahead DS-read / WMMA interleave; the X fragment is prefetched one
// K-step ahead of use.
// ---------------------------------------------------------------------------
__global__ void __launch_bounds__(256) gemm_lstm_fused(
        const f16* __restrict__ Apk, const f16* __restrict__ Xs,
        const float* __restrict__ bias,
        const float* __restrict__ z, const float* __restrict__ zb,
        const float* __restrict__ cold, const float* __restrict__ hold,
        float* __restrict__ out) {
    __shared__ f16 smem[2 * 16 * 512];     // 2 buffers x 16 frags x 1KB = 32 KB

    int tid  = threadIdx.x;
    int lane = tid & 31;
    int wave = tid >> 5;
    int lrow = lane & 15;
    int lgrp = lane >> 4;
    int nb    = blockIdx.x * 128 + wave * 16;
    int tbase = blockIdx.y * 4;
    int col   = nb + lrow;
    const f16* xbase = Xs + (size_t)col * KTOT + lgrp * 16;

    // --- async-copy address setup: 16 KB tile = 1024 chunks of 16B;
    // 256 threads x 4 chunks. chunk cid -> frag f = cid>>6, f = gate*4+tt,
    // global frag (mt = gate*32 + tbase + tt, kt) at ((mt*NKT)+kt)*1024 B.
    uint64_t abase = (uint64_t)Apk;
    uint32_t lds0  = (uint32_t)(uintptr_t)(&smem[0]);
    uint32_t gofs[4], lofs[4];
#pragma unroll
    for (int r = 0; r < 4; ++r) {
        int cid  = tid + r * 256;
        int f    = cid >> 6;               // 0..15
        int coff = (cid & 63) * 16;        // byte offset within frag
        int gate = f >> 2;
        int ttl  = f & 3;
        int mt   = gate * 32 + tbase + ttl;
        gofs[r]  = (uint32_t)(mt * NKT) * 1024u + (uint32_t)coff;  // + kt*1024
        lofs[r]  = (uint32_t)(f * 1024 + coff);                    // + buf*16384
    }

    float zc  = z[col];
    float zbc = zb[col];
    float omz = 1.0f - zc;

    v8f acc[16];
#pragma unroll
    for (int f = 0; f < 16; ++f) acc[f] = (v8f){};

    // prologue: stage kt=0 into buffer 0; prefetch first X fragment
#pragma unroll
    for (int r = 0; r < 4; ++r)
        async_copy_b128(lds0 + lofs[r], abase, gofs[r]);
    v16h xf = *(const v16h*)(xbase);

    uint32_t buf = 0;
#pragma unroll 1
    for (int kt = 0; kt < NKT; ++kt) {
        if (kt + 1 < NKT) {
            // prefetch next K-step's A tile into the other buffer
#pragma unroll
            for (int r = 0; r < 4; ++r)
                async_copy_b128(lds0 + (buf ^ 1) * 16384 + lofs[r],
                                abase, gofs[r] + (uint32_t)(kt + 1) * 1024u);
            asm volatile("s_wait_asynccnt 0x4" ::: "memory");
        } else {
            asm volatile("s_wait_asynccnt 0x0" ::: "memory");
        }
        __syncthreads();                   // whole A tile visible to all waves

        // prefetch next X fragment (global, consumed next iteration)
        v16h xf_next = xf;
        if (kt + 1 < NKT) xf_next = *(const v16h*)(xbase + (kt + 1) * 32);

        const f16* sb = smem + buf * 8192 + lane * 16;
#pragma unroll
        for (int f = 0; f < 16; ++f) {
            const v16h af = *(const v16h*)(sb + f * 512);
            acc[f] = __builtin_amdgcn_wmma_f32_16x16x32_f16(
                false, af, false, xf, (short)0, acc[f], false, false);
        }
        // --- prescribe DS-read / WMMA interleave (3-fragment lookahead) ---
        // 0x100 = DS read, 0x008 = MFMA/WMMA
        __builtin_amdgcn_sched_group_barrier(0x100, 6, 0);   // frags 0..2
#pragma unroll
        for (int s = 0; s < 13; ++s) {
            __builtin_amdgcn_sched_group_barrier(0x008, 1, 0); // wmma f
            __builtin_amdgcn_sched_group_barrier(0x100, 2, 0); // frag f+3
        }
        __builtin_amdgcn_sched_group_barrier(0x008, 3, 0);   // wmma 13..15

        __syncthreads();                   // reads done before buffer reuse
        xf = xf_next;
        buf ^= 1;
    }

    // --- fused LSTM epilogue: acc[gate*4+tt], v8f elem r at row r+lgrp*8 ---
    float* out_h = out;
    float* out_c = out + (size_t)HID * BATCH;
#pragma unroll
    for (int ttl = 0; ttl < 4; ++ttl) {
        int t = tbase + ttl;
#pragma unroll
        for (int r = 0; r < 8; ++r) {
            int row = t * 16 + r + lgrp * 8;          // 0..511
            float fs = acc[0 * 4 + ttl][r] + bias[row];
            float is = acc[1 * 4 + ttl][r] + bias[row + 512];
            float os = acc[2 * 4 + ttl][r] + bias[row + 1024];
            float gs = acc[3 * 4 + ttl][r] + bias[row + 1536];
            float f = sigm(fs);
            float i = sigm(is);
            float o = sigm(os);
            float g = tanh_fast(gs);
            size_t off = (size_t)row * BATCH + col;
            float cprev = cold[off];
            float hprev = hold[off];
            float ig = i * g;
            float cn = zc * ig + omz * (1.0f - zbc) * cprev
                     + omz * zbc * (f * cprev + ig);
            float tc = tanh_fast(cn);
            float hn = zc * o * tc + omz * (1.0f - zbc) * hprev
                     + omz * zbc * o * tc;
            out_h[off] = hn;
            out_c[off] = cn;
        }
    }
}

// ---------------------------------------------------------------------------
// Kernel 4: z_hat row (row 2048) + hard_sigm.
// ---------------------------------------------------------------------------
__global__ void zhat_kernel(const float* __restrict__ h,
                            const float* __restrict__ ht,
                            const float* __restrict__ hb,
                            const float* __restrict__ z,
                            const float* __restrict__ zb,
                            const float* __restrict__ U11,
                            const float* __restrict__ U21,
                            const float* __restrict__ W01,
                            const float* __restrict__ bias,
                            float* __restrict__ out) {
    int b = blockIdx.x * 256 + threadIdx.x;
    const float* w1 = U11 + (size_t)2048 * 512;
    const float* w2 = U21 + (size_t)2048 * 512;
    const float* w3 = W01 + (size_t)2048 * 512;
    float s1 = 0.f, s2 = 0.f, s3 = 0.f;
#pragma unroll 4
    for (int k = 0; k < 512; ++k) {
        s1 = __fmaf_rn(w1[k], h[(size_t)k * BATCH + b], s1);
        s2 = __fmaf_rn(w2[k], ht[(size_t)k * BATCH + b], s2);
        s3 = __fmaf_rn(w3[k], hb[(size_t)k * BATCH + b], s3);
    }
    float fs = (1.0f - z[b]) * s1 + z[b] * s2 + zb[b] * s3 + bias[2048];
    float v  = (fs + 1.0f) * 0.5f;
    out[(size_t)2 * HID * BATCH + b] = fminf(1.0f, fmaxf(0.0f, v));
}

extern "C" void kernel_launch(void* const* d_in, const int* in_sizes, int n_in,
                              void* d_out, int out_size, void* d_ws, size_t ws_size,
                              hipStream_t stream) {
    const float* c        = (const float*)d_in[0];
    const float* h_bottom = (const float*)d_in[1];
    const float* h        = (const float*)d_in[2];
    const float* h_top    = (const float*)d_in[3];
    const float* z        = (const float*)d_in[4];
    const float* z_bottom = (const float*)d_in[5];
    const float* U11      = (const float*)d_in[6];
    const float* U21      = (const float*)d_in[7];
    const float* W01      = (const float*)d_in[8];
    const float* bias     = (const float*)d_in[9];
    float* out = (float*)d_out;

    f16* Apk = (f16*)d_ws;                        // 6,291,456 B
    f16* Xs  = (f16*)((char*)d_ws + 6291456);     // 25,165,824 B

    pack_weights<<<12288, 256, 0, stream>>>(U11, U21, W01, Apk);
    prescale_transpose<<<dim3(NKT, BATCH / 32), 256, 0, stream>>>(
        h, h_top, h_bottom, z, z_bottom, Xs);
    gemm_lstm_fused<<<dim3(BATCH / 128, 8), 256, 0, stream>>>(
        Apk, Xs, bias, z, z_bottom, c, h, out);
    zhat_kernel<<<BATCH / 256, 256, 0, stream>>>(
        h, h_top, h_bottom, z, z_bottom, U11, U21, W01, bias, out);
}
